// MultiHeadAttention_81063212745014
// MI455X (gfx1250) — compile-verified, matrix operand order checked by reading, not run on hardware
//
#include <hip/hip_runtime.h>

#define kB 4
#define kT 2048
#define kC 1024
#define kH 16
#define kD 64

typedef __attribute__((ext_vector_type(16))) _Float16 v16h;
typedef __attribute__((ext_vector_type(8)))  _Float16 v8h;
typedef __attribute__((ext_vector_type(8)))  float    v8f;
typedef __attribute__((ext_vector_type(4)))  unsigned u32x4;
typedef __attribute__((ext_vector_type(8)))  unsigned u32x8;

__device__ __forceinline__ v8f wmma_f16(v16h a, v16h b, v8f c) {
  return __builtin_amdgcn_wmma_f32_16x16x32_f16(false, a, false, b, (short)0, c,
                                                false, false);
}

// LDS byte offset of a __shared__ object (generic addr low 32 bits == LDS offset).
__device__ __forceinline__ unsigned ldsoff(const void* p) {
  return (unsigned)(unsigned long long)p;
}

__device__ __forceinline__ void wait_tensor0() {
#if __has_builtin(__builtin_amdgcn_s_wait_tensorcnt)
  __builtin_amdgcn_s_wait_tensorcnt(0);
#else
  asm volatile("s_wait_tensorcnt 0x0" ::: "memory");
#endif
}

__device__ __forceinline__ void wait_async0() {
#if __has_builtin(__builtin_amdgcn_s_wait_asynccnt)
  __builtin_amdgcn_s_wait_asynccnt(0);
#else
  asm volatile("s_wait_asynccnt 0x0" ::: "memory");
#endif
}

// Async 16B global->LDS copy (per-lane), tracked by ASYNCcnt.
__device__ __forceinline__ void async_g2l_b128(unsigned lds, const void* g) {
  asm volatile("global_load_async_to_lds_b128 %0, %1, off"
               :: "v"(lds), "v"(g) : "memory");
}

// TDM: DMA a 2D tile (tile_d0 x tile_d1 elements of 2 bytes, row stride
// stride0 elements) from global to LDS. Descriptor per CDNA5 ISA sec 8.
__device__ __forceinline__ void tdm_load_2d(unsigned lds, const void* gaddr,
                                            unsigned tile_d0, unsigned tile_d1,
                                            unsigned stride0, unsigned tdim0,
                                            unsigned tdim1) {
  unsigned long long ga = (unsigned long long)gaddr;
  u32x4 g0;
  g0[0] = 1u;                                   // count=1, user mode
  g0[1] = lds;                                  // lds_addr (bytes)
  g0[2] = (unsigned)ga;                         // global_addr[31:0]
  g0[3] = (unsigned)(ga >> 32) | (2u << 30);    // global_addr[56:32] | type=2
  u32x8 g1;
  g1[0] = 0x00010000u;                          // data_size=1 (2 bytes)
  g1[1] = tdim0 << 16;                          // tensor_dim0[15:0]
  g1[2] = (tdim0 >> 16) | (tdim1 << 16);        // tensor_dim0[31:16]|dim1[15:0]
  g1[3] = (tdim1 >> 16) | (tile_d0 << 16);      // tensor_dim1[31:16]|tile_dim0
  g1[4] = tile_d1;                              // tile_dim1 (tile_dim2=0)
  g1[5] = stride0;                              // tensor_dim0_stride[31:0]
  g1[6] = 0;                                    // stride0[47:32]|dim1_stride lo
  g1[7] = 0;
  u32x4 gz = {0, 0, 0, 0};
  asm volatile("tensor_load_to_lds %0, %1, %2, %3"
               :: "s"(g0), "s"(g1), "s"(gz), "s"(gz) : "memory");
}

// Load a 16x32 f16 fragment from row-major storage (leading dim `ld` halves).
// A layout: lane r(<16) holds row r, K {0..7,16..23}; lane r+16 holds row r,
// K {8..15,24..31}. B fragments use the same loader with "row" == column N.
__device__ __forceinline__ v16h load_frag(const _Float16* p, int ld, int lane) {
  const int r = lane & 15, g = lane >> 4;
  const _Float16* row = p + (size_t)r * ld + 8 * g;
  v8h c0 = *(const v8h*)(row);
  v8h c1 = *(const v8h*)(row + 16);
  v16h o;
#pragma unroll
  for (int i = 0; i < 8; ++i) { o[i] = c0[i]; o[i + 8] = c1[i]; }
  return o;
}

// ---------------- prep kernels ----------------
__global__ void cast_f16(const float* __restrict__ s, _Float16* __restrict__ d,
                         int n) {
  for (int i = blockIdx.x * blockDim.x + threadIdx.x; i < n;
       i += gridDim.x * blockDim.x)
    d[i] = (_Float16)s[i];
}

// [H, C, D] fp32 -> [H, D, C] f16  (one thread per element; n == kH*kC*kD)
__global__ void transpose_w(const float* __restrict__ s,
                            _Float16* __restrict__ d) {
  int i = blockIdx.x * blockDim.x + threadIdx.x;
  int h = i >> 16;
  int rem = i & 65535;
  int c = rem >> 6, dd = rem & 63;
  d[(size_t)((h << 6) + dd) * kC + c] = (_Float16)s[i];
}

// ---------------- QKV GEMM (TDM-staged, double buffered) ----------------
// X[8192,1024] * Wt[3*H, D(rows), C(cols)] -> Q,K f16 [bh,t,d], V f16 [bh,d,t]
__global__ __launch_bounds__(256) void gemm_qkv(
    const _Float16* __restrict__ Xh, const _Float16* __restrict__ Wt,
    _Float16* __restrict__ Q, _Float16* __restrict__ K,
    _Float16* __restrict__ Vt) {
  __shared__ __align__(16) _Float16 As[2][256 * 32];
  __shared__ __align__(16) _Float16 Bs[2][64 * 32];
  const int colbase = blockIdx.x * 64;   // over 3*H*D
  const int rowbase = blockIdx.y * 256;  // over B*T
  const int mat = colbase >> 10;
  const int h = (colbase & 1023) >> 6;
  const _Float16* Bsrc = Wt + (size_t)(mat * kH + h) * kD * kC;
  const int tid = threadIdx.x;
  const int wave = tid >> 5;
  const int lane = tid & 31;

  v8f acc[2][4] = {};

  if (wave == 0) {
    tdm_load_2d(ldsoff(&As[0][0]), &Xh[(size_t)rowbase * kC], 32, 256, kC, kC,
                1u << 20);
    tdm_load_2d(ldsoff(&Bs[0][0]), Bsrc, 32, 64, kC, kC, 1u << 20);
  }

  for (int k0 = 0; k0 < kC; k0 += 32) {
    const int cur = (k0 >> 5) & 1;
    if (wave == 0) wait_tensor0();
    __syncthreads();
    if (wave == 0 && k0 + 32 < kC) {
      tdm_load_2d(ldsoff(&As[cur ^ 1][0]),
                  &Xh[(size_t)rowbase * kC + k0 + 32], 32, 256, kC, kC,
                  1u << 20);
      tdm_load_2d(ldsoff(&Bs[cur ^ 1][0]), Bsrc + k0 + 32, 32, 64, kC, kC,
                  1u << 20);
    }
    v16h a0 = load_frag(&As[cur][(wave * 32) * 32], 32, lane);
    v16h a1 = load_frag(&As[cur][(wave * 32 + 16) * 32], 32, lane);
#pragma unroll
    for (int nt = 0; nt < 4; ++nt) {
      v16h b = load_frag(&Bs[cur][nt * 16 * 32], 32, lane);
      acc[0][nt] = wmma_f16(a0, b, acc[0][nt]);
      acc[1][nt] = wmma_f16(a1, b, acc[1][nt]);
    }
  }

  const int g = lane >> 4, ln = lane & 15;
#pragma unroll
  for (int mt = 0; mt < 2; ++mt)
#pragma unroll
    for (int nt = 0; nt < 4; ++nt) {
      int col = nt * 16 + ln;
#pragma unroll
      for (int r = 0; r < 8; ++r) {
        int grow = rowbase + wave * 32 + mt * 16 + r + 8 * g;  // b*T + t
        int b = grow >> 11, t = grow & 2047;
        size_t bh = (size_t)b * kH + h;
        _Float16 val = (_Float16)acc[mt][nt][r];
        if (mat == 0)
          Q[(bh * kT + t) * kD + col] = val;
        else if (mat == 1)
          K[(bh * kT + t) * kD + col] = val;
        else
          Vt[(bh * kD + col) * kT + t] = val;
      }
    }
}

// ---------------- flash attention (async-LDS staged K/V) ----------------
// One wave = 16 query rows; block = 4 waves = 64 rows of one (b,h).
__global__ __launch_bounds__(128) void attn(const _Float16* __restrict__ Q,
                                            const _Float16* __restrict__ K,
                                            const _Float16* __restrict__ Vt,
                                            _Float16* __restrict__ Oc) {
  __shared__ __align__(16) _Float16 Ks[32 * 64];   // [token][d]
  __shared__ __align__(16) _Float16 Vs[64 * 32];   // [d][token]
  __shared__ __align__(16) _Float16 Pls[4 * 16 * 32];
  const int bh = blockIdx.y;
  const int b = bh >> 4, h = bh & 15;
  const int tid = threadIdx.x;
  const int wave = tid >> 5;
  const int lane = tid & 31;
  const int g = lane >> 4, ln = lane & 15;
  const int q0 = blockIdx.x * 64 + wave * 16;

  const _Float16* Qb = Q + ((size_t)bh * kT + q0) * kD;
  v16h aq0 = load_frag(Qb, kD, lane);
  v16h aq1 = load_frag(Qb + 32, kD, lane);

  const unsigned ksO = ldsoff(Ks), vsO = ldsoff(Vs);

  v8f o[4] = {};
  float M[8], L[8];
#pragma unroll
  for (int r = 0; r < 8; ++r) { M[r] = -1e30f; L[r] = 0.f; }
  const float scale = 0.03125f;  // C^-0.5

  for (int j = 0; j < kT; j += 32) {
    // stage K tile (32x64) and V tile (64x32) via async global->LDS
#pragma unroll
    for (int rep = 0; rep < 2; ++rep) {
      int u = tid + rep * 128;
      int kr = u >> 3, kc = u & 7;  // K tile: 32 rows x 8 chunks
      async_g2l_b128(ksO + (unsigned)(kr * 64 + kc * 8) * 2,
                     &K[((size_t)bh * kT + j + kr) * kD + kc * 8]);
      int vr = u >> 2, vc = u & 3;  // V tile: 64 rows x 4 chunks
      async_g2l_b128(vsO + (unsigned)(vr * 32 + vc * 8) * 2,
                     &Vt[((size_t)bh * kD + vr) * kT + j + vc * 8]);
    }
    wait_async0();
    __syncthreads();

    v16h bk00 = load_frag(&Ks[0], 64, lane);
    v16h bk01 = load_frag(&Ks[32], 64, lane);
    v16h bk10 = load_frag(&Ks[16 * 64], 64, lane);
    v16h bk11 = load_frag(&Ks[16 * 64 + 32], 64, lane);

    v8f s0 = {}, s1 = {};
    s0 = wmma_f16(aq0, bk00, s0);
    s0 = wmma_f16(aq1, bk01, s0);
    s1 = wmma_f16(aq0, bk10, s1);
    s1 = wmma_f16(aq1, bk11, s1);

    float corr[8];
#pragma unroll
    for (int r = 0; r < 8; ++r) {
      float v0 = s0[r] * scale, v1 = s1[r] * scale;
      float mx = fmaxf(v0, v1);
#pragma unroll
      for (int m = 1; m < 16; m <<= 1) mx = fmaxf(mx, __shfl_xor(mx, m, 16));
      float Mn = fmaxf(M[r], mx);
      corr[r] = __expf(M[r] - Mn);
      M[r] = Mn;
      float p0 = __expf(v0 - Mn), p1 = __expf(v1 - Mn);
      s0[r] = p0;
      s1[r] = p1;
      float rs = p0 + p1;
#pragma unroll
      for (int m = 1; m < 16; m <<= 1) rs += __shfl_xor(rs, m, 16);
      L[r] = L[r] * corr[r] + rs;
    }
#pragma unroll
    for (int nt = 0; nt < 4; ++nt)
#pragma unroll
      for (int r = 0; r < 8; ++r) o[nt][r] *= corr[r];

    // transpose P (C-layout) -> A-fragment layout via per-wave LDS region
    // (per-wave LDS ops are in-order; no cross-wave sharing -> no barrier)
    _Float16* Pw = &Pls[wave * 16 * 32];
#pragma unroll
    for (int r = 0; r < 8; ++r) {
      int row = r + 8 * g;
      Pw[row * 32 + ln] = (_Float16)s0[r];
      Pw[row * 32 + 16 + ln] = (_Float16)s1[r];
    }
    v16h ap = load_frag(Pw, 32, lane);

#pragma unroll
    for (int nt = 0; nt < 4; ++nt) {
      v16h bv = load_frag(&Vs[nt * 16 * 32], 32, lane);
      o[nt] = wmma_f16(ap, bv, o[nt]);
    }
    __syncthreads();  // protect Ks/Vs from next iteration's async writes
  }

#pragma unroll
  for (int nt = 0; nt < 4; ++nt) {
    int d = nt * 16 + ln;
#pragma unroll
    for (int r = 0; r < 8; ++r) {
      int t = q0 + r + 8 * g;
      Oc[((size_t)b * kT + t) * kC + h * kD + d] = (_Float16)(o[nt][r] / L[r]);
    }
  }
}

// ---------------- output projection (TDM-staged, double buffered) ---------
__global__ __launch_bounds__(256) void gemm_proj(
    const _Float16* __restrict__ Oc, const _Float16* __restrict__ Wph,
    const float* __restrict__ bp, float* __restrict__ Y) {
  __shared__ __align__(16) _Float16 As[2][256 * 32];
  __shared__ __align__(16) _Float16 Bs[2][64 * 32];
  const int colbase = blockIdx.x * 64;
  const int rowbase = blockIdx.y * 256;
  const _Float16* Bsrc = Wph + (size_t)colbase * kC;
  const int tid = threadIdx.x;
  const int wave = tid >> 5;
  const int lane = tid & 31;

  v8f acc[2][4] = {};

  if (wave == 0) {
    tdm_load_2d(ldsoff(&As[0][0]), &Oc[(size_t)rowbase * kC], 32, 256, kC, kC,
                1u << 20);
    tdm_load_2d(ldsoff(&Bs[0][0]), Bsrc, 32, 64, kC, kC, 1u << 20);
  }

  for (int k0 = 0; k0 < kC; k0 += 32) {
    const int cur = (k0 >> 5) & 1;
    if (wave == 0) wait_tensor0();
    __syncthreads();
    if (wave == 0 && k0 + 32 < kC) {
      tdm_load_2d(ldsoff(&As[cur ^ 1][0]),
                  &Oc[(size_t)rowbase * kC + k0 + 32], 32, 256, kC, kC,
                  1u << 20);
      tdm_load_2d(ldsoff(&Bs[cur ^ 1][0]), Bsrc + k0 + 32, 32, 64, kC, kC,
                  1u << 20);
    }
    v16h a0 = load_frag(&As[cur][(wave * 32) * 32], 32, lane);
    v16h a1 = load_frag(&As[cur][(wave * 32 + 16) * 32], 32, lane);
#pragma unroll
    for (int nt = 0; nt < 4; ++nt) {
      v16h b = load_frag(&Bs[cur][nt * 16 * 32], 32, lane);
      acc[0][nt] = wmma_f16(a0, b, acc[0][nt]);
      acc[1][nt] = wmma_f16(a1, b, acc[1][nt]);
    }
  }

  const int g = lane >> 4, ln = lane & 15;
#pragma unroll
  for (int mt = 0; mt < 2; ++mt)
#pragma unroll
    for (int nt = 0; nt < 4; ++nt) {
      int o = colbase + nt * 16 + ln;
      float bias = bp[o];
#pragma unroll
      for (int r = 0; r < 8; ++r) {
        int grow = rowbase + wave * 32 + mt * 16 + r + 8 * g;
        Y[(size_t)grow * kC + o] = acc[mt][nt][r] + bias;
      }
    }
}

extern "C" void kernel_launch(void* const* d_in, const int* in_sizes, int n_in,
                              void* d_out, int out_size, void* d_ws,
                              size_t ws_size, hipStream_t stream) {
  (void)in_sizes; (void)n_in; (void)out_size; (void)ws_size;
  const float* x = (const float*)d_in[0];
  const float* Wq = (const float*)d_in[1];
  const float* Wk = (const float*)d_in[2];
  const float* Wv = (const float*)d_in[3];
  const float* Wp = (const float*)d_in[4];
  const float* bp = (const float*)d_in[5];
  float* out = (float*)d_out;

  char* ws = (char*)d_ws;
  size_t off = 0;
  _Float16* Xh = (_Float16*)(ws + off);  off += (size_t)kB * kT * kC * 2;
  _Float16* Wt = (_Float16*)(ws + off);  off += (size_t)3 * kH * kD * kC * 2;
  _Float16* Wph = (_Float16*)(ws + off); off += (size_t)kC * kC * 2;
  _Float16* Qb = (_Float16*)(ws + off);  off += (size_t)kB * kH * kT * kD * 2;
  _Float16* Kb = (_Float16*)(ws + off);  off += (size_t)kB * kH * kT * kD * 2;
  _Float16* Vtb = (_Float16*)(ws + off); off += (size_t)kB * kH * kT * kD * 2;
  _Float16* Ocb = (_Float16*)(ws + off); off += (size_t)kB * kT * kC * 2;

  cast_f16<<<4096, 256, 0, stream>>>(x, Xh, kB * kT * kC);
  transpose_w<<<4096, 256, 0, stream>>>(Wq, Wt);
  transpose_w<<<4096, 256, 0, stream>>>(Wk, Wt + (size_t)kH * kD * kC);
  transpose_w<<<4096, 256, 0, stream>>>(Wv, Wt + (size_t)2 * kH * kD * kC);
  cast_f16<<<2048, 256, 0, stream>>>(Wp, Wph, kC * kC);

  gemm_qkv<<<dim3(48, 32), 256, 0, stream>>>(Xh, Wt, Qb, Kb, Vtb);
  attn<<<dim3(kT / 64, kB * kH), 128, 0, stream>>>(Qb, Kb, Vtb, Ocb);
  gemm_proj<<<dim3(kC / 64, 32), 256, 0, stream>>>(Ocb, Wph, bp, out);
}
